// PredictTime_66881230733364
// MI455X (gfx1250) — matrix operand order repaired and checked by reading, not compiled, not run on hardware
//
#include <hip/hip_runtime.h>
#include <hip/hip_bf16.h>

// ---------------------------------------------------------------------------
// 2-layer LSTM (T=512, B=256, IN=64, H=1024, OUT=1) for gfx1250 (MI455X).
// bf16 WMMA GEMMs with fused LSTM pointwise epilogue; 3 kernels per timestep.
// ---------------------------------------------------------------------------

#define T_STEPS 512
#define BATCH   256
#define IN_DIM  64
#define HID     1024

typedef __attribute__((ext_vector_type(16))) __bf16 v16bf;
typedef __attribute__((ext_vector_type(8)))  float  v8f;

union FragU {
    v16bf v;
    struct { uint4 a, b; } u;
};

__device__ __forceinline__ unsigned short f2bf(float f) {
    unsigned u = __float_as_uint(f);
    unsigned r = (u + 0x7FFFu + ((u >> 16) & 1u)) >> 16;   // RNE
    return (unsigned short)r;
}
__device__ __forceinline__ float bf2f(unsigned short h) {
    return __uint_as_float((unsigned)h << 16);
}
__device__ __forceinline__ float sigf(float x) {
    return 1.0f / (1.0f + __expf(-x));
}

// Load a 16x32 bf16 A fragment: two 16B chunks per lane.
__device__ __forceinline__ v16bf load_afrag(const unsigned short* row, int hi) {
    FragU f;
    const uint4* p0 = (const uint4*)(row + hi * 8);        // K = hi*8 .. hi*8+7
    const uint4* p1 = (const uint4*)(row + 16 + hi * 8);   // K = 16+hi*8 ..
    f.u.a = *p0;
    f.u.b = *p1;
    return f.v;
}

// Load a 32x16 bf16 B fragment: two 16B chunks per lane at p and p+16 (K-major row).
__device__ __forceinline__ v16bf load_bfrag(const unsigned short* p) {
    FragU f;
    f.u.a = *(const uint4*)(p);
    f.u.b = *(const uint4*)(p + 16);
    return f.v;
}

// ---------------------------------------------------------------------------
// Fused GEMM + LSTM cell.
//   gates[B,4H] = [A1|A2] @ W^T + bias ; then c,h update.
// W layout: [4 gates][HID rows][K] row-major (K contiguous), bf16.
// A split in two K-segments (x|h or h1|h2), each bf16 row-major.
// Grid: (HID/16, BATCH/128), block 256 threads = 8 waves.
// Wave w owns rows [blockIdx.y*128 + w*16, +16), hidden units [blockIdx.x*16,+16),
// all four gates -> pointwise update entirely local to the wave.
// ---------------------------------------------------------------------------
__global__ void lstm_gemm_kernel(
    const unsigned short* __restrict__ A1, int lda1, int k1steps,
    const unsigned short* __restrict__ A2, int lda2, int k2steps,
    const unsigned short* __restrict__ W,  int K,
    const float* __restrict__ bias,
    float* __restrict__ c,
    unsigned short* __restrict__ hout)
{
    const int lane = threadIdx.x & 31;
    const int wave = threadIdx.x >> 5;
    const int nl   = lane & 15;
    const int hi   = lane >> 4;

    const int j0 = blockIdx.x * 16;           // hidden-unit tile
    const int m0 = blockIdx.y * 128 + wave * 16;

    // Bias init: C element r of lane L is (M = 8*hi + r, N = nl) -> bias
    // depends only on N.
    v8f acc[4];
#pragma unroll
    for (int g = 0; g < 4; ++g) {
        float bv = bias[g * HID + j0 + nl];
#pragma unroll
        for (int r = 0; r < 8; ++r) acc[g][r] = bv;
    }

    // Per-gate weight row pointers for this lane (row = output unit j0+nl).
    const unsigned short* wrow[4];
#pragma unroll
    for (int g = 0; g < 4; ++g)
        wrow[g] = W + (size_t)(g * HID + j0 + nl) * K + hi * 8;

    const unsigned short* arow1 = A1 + (size_t)(m0 + nl) * lda1;
    const unsigned short* arow2 = A2 + (size_t)(m0 + nl) * lda2;

    int kglob = 0;
#pragma unroll 2
    for (int kt = 0; kt < k1steps; ++kt, kglob += 32) {
        v16bf af = load_afrag(arow1 + kt * 32, hi);
        __builtin_prefetch(arow1 + kt * 32 + 32, 0, 1);
#pragma unroll
        for (int g = 0; g < 4; ++g) {
            v16bf bf = load_bfrag(wrow[g] + kglob);
            acc[g] = __builtin_amdgcn_wmma_f32_16x16x32_bf16(
                false, af, false, bf, (short)0, acc[g], false, false);
        }
    }
#pragma unroll 2
    for (int kt = 0; kt < k2steps; ++kt, kglob += 32) {
        v16bf af = load_afrag(arow2 + kt * 32, hi);
        __builtin_prefetch(arow2 + kt * 32 + 32, 0, 1);
#pragma unroll
        for (int g = 0; g < 4; ++g) {
            v16bf bf = load_bfrag(wrow[g] + kglob);
            acc[g] = __builtin_amdgcn_wmma_f32_16x16x32_bf16(
                false, af, false, bf, (short)0, acc[g], false, false);
        }
    }

    // Fused LSTM pointwise epilogue (PyTorch gate order i,f,g,o).
    const int ncol = j0 + nl;
#pragma unroll
    for (int r = 0; r < 8; ++r) {
        int m = m0 + hi * 8 + r;
        size_t idx = (size_t)m * HID + ncol;
        float iv = acc[0][r];
        float fv = acc[1][r];
        float gv = acc[2][r];
        float ov = acc[3][r];
        float cold = c[idx];
        float cn = sigf(fv) * cold + sigf(iv) * tanhf(gv);
        c[idx] = cn;
        float h = sigf(ov) * tanhf(cn);
        hout[idx] = f2bf(h);
    }
}

// ---------------------------------------------------------------------------
// y[b] = sigmoid(dot(h2[b,:], W2) + b2). One block per batch row.
// ---------------------------------------------------------------------------
__global__ void lstm_out_kernel(const unsigned short* __restrict__ h2,
                                const float* __restrict__ W2,
                                const float* __restrict__ b2,
                                float* __restrict__ y)
{
    __shared__ float red[256];
    const int b = blockIdx.x;
    const int tid = threadIdx.x;
    const unsigned short* hp = h2 + (size_t)b * HID;

    float s = 0.f;
#pragma unroll
    for (int i = 0; i < 4; ++i) {
        int k = tid * 4 + i;
        s += bf2f(hp[k]) * W2[k];
    }
    red[tid] = s;
    __syncthreads();
#pragma unroll
    for (int off = 128; off > 0; off >>= 1) {
        if (tid < off) red[tid] += red[tid + off];
        __syncthreads();
    }
    if (tid == 0) y[b] = sigf(red[0] + b2[0]);
}

// ---------------------------------------------------------------------------
// Prep kernels (run every call; deterministic).
// ---------------------------------------------------------------------------
// dst[n*K + k] = bf16( k < k1 ? ih[n*k1 + k] : hh[n*(K-k1) + (k-k1)] )
__global__ void repack_weights_kernel(const float* __restrict__ ih,
                                      const float* __restrict__ hh,
                                      int k1, int K,
                                      unsigned short* __restrict__ dst,
                                      long total)
{
    long stride = (long)gridDim.x * blockDim.x;
    for (long i = (long)blockIdx.x * blockDim.x + threadIdx.x; i < total; i += stride) {
        long n = i / K;
        int  k = (int)(i - n * K);
        float v = (k < k1) ? ih[n * k1 + k] : hh[n * (long)(K - k1) + (k - k1)];
        dst[i] = f2bf(v);
    }
}

__global__ void combine_bias_kernel(const float* __restrict__ bih1,
                                    const float* __restrict__ bhh1,
                                    const float* __restrict__ bih2,
                                    const float* __restrict__ bhh2,
                                    float* __restrict__ bb1,
                                    float* __restrict__ bb2)
{
    int n = blockIdx.x * blockDim.x + threadIdx.x;
    if (n < 4 * HID) {
        bb1[n] = bih1[n] + bhh1[n];
        bb2[n] = bih2[n] + bhh2[n];
    }
}

__global__ void cvt_x_kernel(const float* __restrict__ x,
                             unsigned short* __restrict__ xbf,
                             long total)
{
    long stride = (long)gridDim.x * blockDim.x;
    for (long i = (long)blockIdx.x * blockDim.x + threadIdx.x; i < total; i += stride)
        xbf[i] = f2bf(x[i]);
}

// ---------------------------------------------------------------------------
extern "C" void kernel_launch(void* const* d_in, const int* in_sizes, int n_in,
                              void* d_out, int out_size, void* d_ws, size_t ws_size,
                              hipStream_t stream)
{
    const float* x     = (const float*)d_in[0];
    const float* W_ih1 = (const float*)d_in[1];
    const float* W_hh1 = (const float*)d_in[2];
    const float* b_ih1 = (const float*)d_in[3];
    const float* b_hh1 = (const float*)d_in[4];
    const float* W_ih2 = (const float*)d_in[5];
    const float* W_hh2 = (const float*)d_in[6];
    const float* b_ih2 = (const float*)d_in[7];
    const float* b_hh2 = (const float*)d_in[8];
    const float* W2    = (const float*)d_in[9];
    const float* b2    = (const float*)d_in[10];
    float* out = (float*)d_out;

    // ---- carve workspace (256B aligned) ----
    char* w = (char*)d_ws;
    size_t o = 0;
    auto carve = [&](size_t bytes) -> void* {
        void* p = w + o;
        o = (o + bytes + 255) & ~(size_t)255;
        return p;
    };
    const int K1 = IN_DIM + HID;   // 1088
    const int K2 = 2 * HID;        // 2048

    unsigned short* Wb1 = (unsigned short*)carve((size_t)4 * HID * K1 * 2);
    unsigned short* Wb2 = (unsigned short*)carve((size_t)4 * HID * K2 * 2);
    unsigned short* Xbf = (unsigned short*)carve((size_t)T_STEPS * BATCH * IN_DIM * 2);
    float* bb1 = (float*)carve(4 * HID * sizeof(float));
    float* bb2 = (float*)carve(4 * HID * sizeof(float));

    char* zero_base = w + o;
    float* c1 = (float*)carve((size_t)BATCH * HID * sizeof(float));
    float* c2 = (float*)carve((size_t)BATCH * HID * sizeof(float));
    unsigned short* H1[2], * H2[2];
    H1[0] = (unsigned short*)carve((size_t)BATCH * HID * 2);
    H1[1] = (unsigned short*)carve((size_t)BATCH * HID * 2);
    H2[0] = (unsigned short*)carve((size_t)BATCH * HID * 2);
    H2[1] = (unsigned short*)carve((size_t)BATCH * HID * 2);
    size_t zero_bytes = (size_t)((w + o) - zero_base);

    // ---- state init + weight/bias/x prep (every call; graph-capturable) ----
    hipMemsetAsync(zero_base, 0, zero_bytes, stream);
    repack_weights_kernel<<<4096, 256, 0, stream>>>(W_ih1, W_hh1, IN_DIM, K1, Wb1,
                                                    (long)4 * HID * K1);
    repack_weights_kernel<<<4096, 256, 0, stream>>>(W_ih2, W_hh2, HID, K2, Wb2,
                                                    (long)4 * HID * K2);
    combine_bias_kernel<<<16, 256, 0, stream>>>(b_ih1, b_hh1, b_ih2, b_hh2, bb1, bb2);
    cvt_x_kernel<<<8192, 256, 0, stream>>>(x, Xbf, (long)T_STEPS * BATCH * IN_DIM);

    // ---- time loop: 2 fused GEMM+LSTM kernels + 1 output kernel per step ----
    dim3 gg(HID / 16, BATCH / 128);   // 64 x 2 workgroups, 256 threads each
    for (int t = 0; t < T_STEPS; ++t) {
        unsigned short* h1cur  = H1[t & 1];
        unsigned short* h1prev = H1[(t & 1) ^ 1];
        unsigned short* h2cur  = H2[t & 1];
        unsigned short* h2prev = H2[(t & 1) ^ 1];

        // Layer 1: A = [x_t (K=64) | h1_{t-1} (K=1024)]
        lstm_gemm_kernel<<<gg, 256, 0, stream>>>(
            Xbf + (size_t)t * BATCH * IN_DIM, IN_DIM, IN_DIM / 32,
            h1prev, HID, HID / 32,
            Wb1, K1, bb1, c1, h1cur);

        // Layer 2: A = [h1_t (K=1024) | h2_{t-1} (K=1024)]
        lstm_gemm_kernel<<<gg, 256, 0, stream>>>(
            h1cur, HID, HID / 32,
            h2prev, HID, HID / 32,
            Wb2, K2, bb2, c2, h2cur);

        // y[t, b] = sigmoid(h2_t[b] . W2 + b2)
        lstm_out_kernel<<<BATCH, 256, 0, stream>>>(h2cur, W2, b2, out + (size_t)t * BATCH);
    }
}
